// ModelGNNMultiLayer_31361851196080
// MI455X (gfx1250) — matrix operand order repaired and checked by reading, not compiled, not run on hardware
//
#include <hip/hip_runtime.h>

// Fused GNN (2x GraphConv + JK-cat + 3-layer MLP head) for MI455X (gfx1250).
// All GEMMs via v_wmma_f32_16x16x32_f16; one wave processes one scene tile
// (M = 16 agents). adj@(h@W) associativity folds neighbor aggregation into
// one extra WMMA per N-tile. Weights pre-packed into B-fragment layout in LDS.
// B-fragment loads are software-pipelined (1-deep prefetch) so ds_load latency
// overlaps WMMA execution instead of serializing load->wait->wmma.

typedef _Float16 half_t;
typedef __attribute__((ext_vector_type(16))) _Float16 v16h;
typedef __attribute__((ext_vector_type(8)))  float    v8f;

#define WAVES_PER_BLOCK 8
#define SCENES_PER_WAVE 4
#define NAG 16
#define CIN 64
#define CH  128
#define COMM_R2 64.0f

// Packed weight fragments: one (K=32)x16 B-fragment = 32 lanes * 16 halfs = 512 halfs.
#define FRAG_HALFS 512
#define FB_REL1  0     // 64x128  -> 2*8  = 16 frags
#define FB_ROOT1 16    // 64x128  -> 16 frags
#define FB_REL2  32    // 128x128 -> 4*8  = 32 frags
#define FB_ROOT2 64    // 128x128 -> 32 frags
#define FB_H1    96    // 256x64  -> 8*4  = 32 frags
#define FB_H2    128   // 64x32   -> 2*2  = 4 frags
#define FB_H3    132   // 32x16   -> 1 frag
#define NFRAGS   133
#define W_TOTAL_HALFS (NFRAGS * FRAG_HALFS)   // 68096 halfs = 136192 B

// Wave-local LDS ordering fence (DS ops are in-order per wave; this stops the
// compiler reordering C-level LDS accesses and drains DScnt).
#define DSFENCE() asm volatile("s_wait_dscnt 0" ::: "memory")

#define WMMA_F16(a, b, c) \
  __builtin_amdgcn_wmma_f32_16x16x32_f16(false, (a), false, (b), (short)0, (c), false, false)

// ---- fragment loaders -------------------------------------------------------

// B-fragment from packed LDS weights: lane reads its 16 contiguous halfs.
__device__ inline v16h bfrag_w(const half_t* wlds, int fragIdx, int lane) {
  const uint4* p = (const uint4*)(wlds + (size_t)fragIdx * FRAG_HALFS + lane * 16);
  union { v16h v; uint4 u[2]; } r;
  r.u[0] = p[0];
  r.u[1] = p[1];
  return r.v;
}

// B-fragment of t (16 x CH, K = agent dim padded 16->32) from transposed LDS
// buffer tT[col][row]. halfs 0..7 -> K = 8g+0..7; halfs 8..15 (K=16..31) -> zeros.
__device__ inline v16h bfrag_t(const half_t* tT, const half_t* zpad,
                               int nt, int l16, int g) {
  const half_t* p = tT + (nt * 16 + l16) * 16 + 8 * g;
  const uint4* q = (const uint4*)p;
  union { v16h v; uint4 u[2]; } r;
  r.u[0] = q[0];
  r.u[1] = ((const uint4*)zpad)[0];
  return r.v;
}

// A-fragment (16x32, f16) from a row-major LDS activation buffer.
// Lane layout: row = lane%16; halfs 0..7 -> K = kbase+8g+0..7,
// halfs 8..15 -> K = kbase+16+8g+0..7.
__device__ inline v16h afrag_lds(const half_t* m, int strideH, int row,
                                 int kbase, int g) {
  const half_t* p = m + row * strideH + kbase + 8 * g;
  union { v16h v; uint4 u[2]; } r;
  r.u[0] = *(const uint4*)(p);
  r.u[1] = *(const uint4*)(p + 16);
  return r.v;
}

// ---- software-pipelined GEMM phases ----------------------------------------

// t = A(16xK) @ W(KxNT*16); per N-tile convert to f16 and store transposed
// into tT[col][row]. B fragments prefetched 1 iteration ahead.
template<int KT>
__device__ inline void phaseT(const v16h* A, const half_t* wlds, int fragBase,
                              half_t* tT, int lane, int l16, int g) {
  constexpr int NT = 8;
  v16h b = bfrag_w(wlds, fragBase, lane);            // (kt=0, nt=0)
#pragma unroll
  for (int nt = 0; nt < NT; ++nt) {
    v8f t = {};
#pragma unroll
    for (int kt = 0; kt < KT; ++kt) {
      const int it = nt * KT + kt;
      v16h bn = b;
      if (it + 1 < NT * KT) {
        const int n2 = (it + 1) / KT, k2 = (it + 1) % KT;
        bn = bfrag_w(wlds, fragBase + k2 * NT + n2, lane);
      }
      t = WMMA_F16(A[kt], b, t);
      b = bn;
    }
    union { half_t h[8]; uint4 u; } pk;
#pragma unroll
    for (int r = 0; r < 8; ++r) pk.h[r] = (half_t)t[r];
    *(uint4*)(tT + (nt * 16 + l16) * 16 + 8 * g) = pk.u;
  }
}

// acc[nt] += A(16xK) @ W(KxNT*16), pipelined B-fragment loads.
template<int NT, int KT>
__device__ inline void phaseAcc(v8f (&acc)[NT], const v16h* A, const half_t* wlds,
                                int fragBase, int lane) {
  v16h b = bfrag_w(wlds, fragBase, lane);            // (kt=0, nt=0)
#pragma unroll
  for (int nt = 0; nt < NT; ++nt) {
#pragma unroll
    for (int kt = 0; kt < KT; ++kt) {
      const int it = nt * KT + kt;
      v16h bn = b;
      if (it + 1 < NT * KT) {
        const int n2 = (it + 1) / KT, k2 = (it + 1) % KT;
        bn = bfrag_w(wlds, fragBase + k2 * NT + n2, lane);
      }
      acc[nt] = WMMA_F16(A[kt], b, acc[nt]);
      b = bn;
    }
  }
}

// acc[nt] += adj @ t, B-fragments (columns of t) prefetched 1 tile ahead.
template<int NT>
__device__ inline void phaseAdj(v8f (&acc)[NT], v16h adjA, const half_t* tT,
                                const half_t* zpad, int l16, int g) {
  v16h bt = bfrag_t(tT, zpad, 0, l16, g);
#pragma unroll
  for (int nt = 0; nt < NT; ++nt) {
    v16h btn = bt;
    if (nt + 1 < NT) btn = bfrag_t(tT, zpad, nt + 1, l16, g);
    acc[nt] = WMMA_F16(adjA, bt, acc[nt]);
    bt = btn;
  }
}

// bias + ReLU + row-major f16 store into LDS activation buffer.
template<int NT>
__device__ inline void biasReluStore(v8f (&acc)[NT], const float* __restrict__ bias,
                                     half_t* act, int strideH, int l16, int g) {
#pragma unroll
  for (int nt = 0; nt < NT; ++nt) {
    const float bv = bias[nt * 16 + l16];
#pragma unroll
    for (int r = 0; r < 8; ++r) {
      float v = acc[nt][r] + bv;
      act[(r + 8 * g) * strideH + nt * 16 + l16] = (half_t)(v > 0.0f ? v : 0.0f);
    }
  }
}

// ---- one-time per-block weight packing (global f32 -> LDS f16 B-frag layout)
__device__ inline void packW(const float* __restrict__ W, half_t* wlds,
                             int fragBase, int K, int N, int tid) {
  const int ntiles = N >> 4;
  const int nf = (K >> 5) * ntiles;
  for (int idx = tid; idx < nf * 32; idx += WAVES_PER_BLOCK * 32) {
    int frag = idx >> 5, lane = idx & 31;
    int kt = frag / ntiles, nt = frag - kt * ntiles;
    int gg = lane >> 4, ll = lane & 15;
    int n  = nt * 16 + ll;
    int kb = kt * 32 + gg * 8;           // B-frag: K = 8*g + i (i<8), 16 + 8*g + (i-8) (i>=8)
    union { half_t h[16]; uint4 u[2]; } pk;
#pragma unroll
    for (int i = 0; i < 16; ++i) pk.h[i] = (half_t)W[(size_t)(kb + i + (i >= 8 ? 8 : 0)) * N + n];
    uint4* d = (uint4*)(wlds + (size_t)(fragBase + frag) * FRAG_HALFS + lane * 16);
    d[0] = pk.u[0];
    d[1] = pk.u[1];
  }
}

// ---- fused kernel -----------------------------------------------------------

__global__ void __launch_bounds__(WAVES_PER_BLOCK * 32)
gnn_fused_kernel(const float* __restrict__ x, const float* __restrict__ pos,
                 const float* __restrict__ W_rel1, const float* __restrict__ b_rel1,
                 const float* __restrict__ W_root1,
                 const float* __restrict__ W_rel2, const float* __restrict__ b_rel2,
                 const float* __restrict__ W_root2,
                 const float* __restrict__ Wh1, const float* __restrict__ bh1,
                 const float* __restrict__ Wh2, const float* __restrict__ bh2,
                 const float* __restrict__ Wh3, const float* __restrict__ bh3,
                 float* __restrict__ out, int nScenes)
{
  extern __shared__ uint4 smem4[];
  half_t* wlds = reinterpret_cast<half_t*>(smem4);
  const int tid = threadIdx.x;

  // Stage all weights into LDS in WMMA B-fragment layout (once per block).
  packW(W_rel1,  wlds, FB_REL1,  CIN,    CH, tid);
  packW(W_root1, wlds, FB_ROOT1, CIN,    CH, tid);
  packW(W_rel2,  wlds, FB_REL2,  CH,     CH, tid);
  packW(W_root2, wlds, FB_ROOT2, CH,     CH, tid);
  packW(Wh1,     wlds, FB_H1,    2 * CH, 64, tid);
  packW(Wh2,     wlds, FB_H2,    64,     32, tid);
  packW(Wh3,     wlds, FB_H3,    32,     16, tid);
  half_t* zpad = wlds + W_TOTAL_HALFS;   // 32 zero halfs for K-padding lanes
  if (tid < 4) {
    uint4 z; z.x = 0u; z.y = 0u; z.z = 0u; z.w = 0u;
    ((uint4*)zpad)[tid] = z;
  }
  __syncthreads();

  const int wave = tid >> 5, lane = tid & 31;
  const int g = lane >> 4, l16 = lane & 15;
  half_t* tT  = wlds + W_TOTAL_HALFS + 32 + wave * 4096;  // 2048 halfs: t transposed [col][row]
  half_t* act = tT + 2048;                                // 2048 halfs: row-major activations

  const int scene0 = blockIdx.x * (WAVES_PER_BLOCK * SCENES_PER_WAVE)
                   + wave * SCENES_PER_WAVE;

  for (int si = 0; si < SCENES_PER_WAVE; ++si) {
    const int scene = scene0 + si;
    if (scene >= nScenes) break;

    // Prefetch next scene's inputs into cache (global_prefetch_b8).
    if (si + 1 < SCENES_PER_WAVE && scene + 1 < nScenes) {
      __builtin_prefetch(x + (size_t)((scene + 1) * NAG + l16) * CIN, 0, 0);
      __builtin_prefetch(pos + (size_t)(scene + 1) * NAG * 2, 0, 0);
    }

    // ---- adjacency as f16 A-fragment (16x16 padded to 16x32, K>=16 zero) ----
    const float* ps = pos + (size_t)scene * NAG * 2;
    const float px = ps[l16 * 2 + 0], py = ps[l16 * 2 + 1];
    v16h adjA = {};
#pragma unroll
    for (int i = 0; i < 8; ++i) {
      int c = 8 * g + i;                               // K = 8g + i for halfs 0..7
      float dx = px - ps[2 * c], dy = py - ps[2 * c + 1];
      float ok = ((dx * dx + dy * dy) <= COMM_R2 && c != l16) ? 1.0f : 0.0f;
      adjA[i] = (half_t)ok;
    }

    // ---- h (input features) as A-fragments, straight from global ----
    const float* hrow = x + (size_t)(scene * NAG + l16) * CIN;
    v16h hA[2];
#pragma unroll
    for (int kt = 0; kt < 2; ++kt) {
      const float* p = hrow + kt * 32 + 8 * g;
      v16h v;
#pragma unroll
      for (int i = 0; i < 8; ++i) v[i] = (half_t)p[i];
#pragma unroll
      for (int i = 0; i < 8; ++i) v[8 + i] = (half_t)p[16 + i];
      hA[kt] = v;
    }

    v8f zero = {};

    // ======================= layer 1 =======================
    phaseT<2>(hA, wlds, FB_REL1, tT, lane, l16, g);     // t1 = h @ W_rel1 -> tT
    v8f acc[8];
#pragma unroll
    for (int nt = 0; nt < 8; ++nt) acc[nt] = zero;
    phaseAcc<8, 2>(acc, hA, wlds, FB_ROOT1, lane);      // += h @ W_root1 (no tT dep)
    DSFENCE();                                          // tT stores visible
    phaseAdj<8>(acc, adjA, tT, zpad, l16, g);           // += adj @ t1
    biasReluStore<8>(acc, b_rel1, act, CH, l16, g);     // x1 -> act
    DSFENCE();
    v16h x1A[4];
#pragma unroll
    for (int kt = 0; kt < 4; ++kt) x1A[kt] = afrag_lds(act, CH, l16, kt * 32, g);

    // ======================= layer 2 =======================
    phaseT<4>(x1A, wlds, FB_REL2, tT, lane, l16, g);    // t2 = x1 @ W_rel2 -> tT
#pragma unroll
    for (int nt = 0; nt < 8; ++nt) acc[nt] = zero;
    phaseAcc<8, 4>(acc, x1A, wlds, FB_ROOT2, lane);     // += x1 @ W_root2
    DSFENCE();
    phaseAdj<8>(acc, adjA, tT, zpad, l16, g);           // += adj @ t2
    biasReluStore<8>(acc, b_rel2, act, CH, l16, g);     // x2 -> act
    DSFENCE();
    v16h x2A[4];
#pragma unroll
    for (int kt = 0; kt < 4; ++kt) x2A[kt] = afrag_lds(act, CH, l16, kt * 32, g);

    // ======================= head =======================
    // h1 = relu([x1,x2] @ Wh1 + bh1)       (16x256 @ 256x64)
    v16h hcatA[8];
#pragma unroll
    for (int kt = 0; kt < 4; ++kt) { hcatA[kt] = x1A[kt]; hcatA[4 + kt] = x2A[kt]; }
    v8f acc4[4];
#pragma unroll
    for (int nt = 0; nt < 4; ++nt) acc4[nt] = zero;
    {
      // pipelined 16x256 @ 256x64 (frag layout kt*4 + nt)
      v16h b = bfrag_w(wlds, FB_H1, lane);
#pragma unroll
      for (int nt = 0; nt < 4; ++nt) {
#pragma unroll
        for (int kt = 0; kt < 8; ++kt) {
          const int it = nt * 8 + kt;
          v16h bn = b;
          if (it + 1 < 32) {
            const int n2 = (it + 1) / 8, k2 = (it + 1) % 8;
            bn = bfrag_w(wlds, FB_H1 + k2 * 4 + n2, lane);
          }
          acc4[nt] = WMMA_F16(hcatA[kt], b, acc4[nt]);
          b = bn;
        }
      }
    }
    biasReluStore<4>(acc4, bh1, act, 64, l16, g);       // h1 -> act (16x64)
    DSFENCE();
    v16h h1A[2];
#pragma unroll
    for (int kt = 0; kt < 2; ++kt) h1A[kt] = afrag_lds(act, 64, l16, kt * 32, g);

    // h2 = relu(h1 @ Wh2 + bh2)            (16x64 @ 64x32)
    v8f acc2[2];
#pragma unroll
    for (int nt = 0; nt < 2; ++nt) acc2[nt] = zero;
    {
      v16h b = bfrag_w(wlds, FB_H2, lane);
#pragma unroll
      for (int nt = 0; nt < 2; ++nt) {
#pragma unroll
        for (int kt = 0; kt < 2; ++kt) {
          const int it = nt * 2 + kt;
          v16h bn = b;
          if (it + 1 < 4) {
            const int n2 = (it + 1) / 2, k2 = (it + 1) % 2;
            bn = bfrag_w(wlds, FB_H2 + k2 * 2 + n2, lane);
          }
          acc2[nt] = WMMA_F16(h1A[kt], b, acc2[nt]);
          b = bn;
        }
      }
    }
    biasReluStore<2>(acc2, bh2, act, 32, l16, g);       // h2 -> act (16x32)
    DSFENCE();
    v16h h2A = afrag_lds(act, 32, l16, 0, g);

    // out = h2 @ Wh3 + bh3                 (16x32 @ 32x16), no ReLU
    {
      v8f accO = zero;
      accO = WMMA_F16(h2A, bfrag_w(wlds, FB_H3, lane), accO);
      const float bv = bh3[l16];
      float* orow = out + (size_t)scene * NAG * 16;
#pragma unroll
      for (int r = 0; r < 8; ++r)
        orow[(r + 8 * g) * 16 + l16] = accO[r] + bv;
    }
    DSFENCE();
  }
}

// ---- launcher ---------------------------------------------------------------

extern "C" void kernel_launch(void* const* d_in, const int* in_sizes, int n_in,
                              void* d_out, int out_size, void* d_ws, size_t ws_size,
                              hipStream_t stream) {
  (void)n_in; (void)out_size; (void)d_ws; (void)ws_size;
  const float* x       = (const float*)d_in[0];
  const float* pos     = (const float*)d_in[1];
  const float* W_rel1  = (const float*)d_in[2];
  const float* b_rel1  = (const float*)d_in[3];
  const float* W_root1 = (const float*)d_in[4];
  const float* W_rel2  = (const float*)d_in[5];
  const float* b_rel2  = (const float*)d_in[6];
  const float* W_root2 = (const float*)d_in[7];
  const float* Wh1     = (const float*)d_in[8];
  const float* bh1     = (const float*)d_in[9];
  const float* Wh2     = (const float*)d_in[10];
  const float* bh2     = (const float*)d_in[11];
  const float* Wh3     = (const float*)d_in[12];
  const float* bh3     = (const float*)d_in[13];
  float* out = (float*)d_out;

  const int nNodes  = in_sizes[0] / CIN;
  const int nScenes = nNodes / NAG;
  const int scenesPerBlock = WAVES_PER_BLOCK * SCENES_PER_WAVE;
  const int grid = (nScenes + scenesPerBlock - 1) / scenesPerBlock;
  const size_t smem =
      (size_t)(W_TOTAL_HALFS + 32 + WAVES_PER_BLOCK * 4096) * sizeof(half_t);

  gnn_fused_kernel<<<grid, WAVES_PER_BLOCK * 32, smem, stream>>>(
      x, pos, W_rel1, b_rel1, W_root1, W_rel2, b_rel2, W_root2,
      Wh1, bh1, Wh2, bh2, Wh3, bh3, out, nScenes);
}
